// MultiGRUDecoder_32976758899057
// MI455X (gfx1250) — matrix operand order repaired and checked by reading, not compiled, not run on hardware
//
#include <hip/hip_runtime.h>

// ---------------------------------------------------------------------------
// Fused 3-layer GRU decoder for MI455X (gfx1250, wave32, WMMA bf16->f32).
// One persistent kernel runs all 72 timesteps; weights live in L2 as packed
// bf16 WMMA B-fragments; hidden states live in LDS in WMMA A-layout.
// ---------------------------------------------------------------------------

typedef __attribute__((ext_vector_type(16))) __bf16 v16bf;
typedef __attribute__((ext_vector_type(8)))  __bf16 v8bf;
typedef __attribute__((ext_vector_type(8)))  float  v8f;

namespace {
constexpr int B_  = 2048;
constexpr int T_  = 72;
constexpr int I_  = 200;
constexpr int H_  = 200;
constexpr int O_  = 53;
constexpr int KP  = 224;          // K padded to 7 * 32
constexpr int KT  = 7;            // k-steps of 32
constexpr int NTG = 13;           // 16-wide n-tiles per gate (208 cols)
constexpr int NPG = 208;
constexpr int NTO = 4;            // out-proj n-tiles (64 cols for O=53)
constexpr int FRAG_SH = 512;      // ushorts per 32x16 bf16 B-fragment (1KB)
constexpr int BUF_SH  = 16 * KP;  // one LDS h-buffer (16 rows x 224) in ushorts

constexpr int GSH      = 3 * NTG * KT * FRAG_SH;       // gated matrix, ushorts
constexpr int WIN_OFF  = 0;
constexpr int WIH1_OFF = WIN_OFF  + NTG * KT * FRAG_SH;
constexpr int WHH1_OFF = WIH1_OFF + GSH;
constexpr int WIH2_OFF = WHH1_OFF + GSH;
constexpr int WHH2_OFF = WIH2_OFF + GSH;
constexpr int WIH3_OFF = WHH2_OFF + GSH;
constexpr int WHH3_OFF = WIH3_OFF + GSH;
constexpr int WOUT_OFF = WHH3_OFF + GSH;
constexpr size_t XW1_OFF_BYTES = (size_t)(WOUT_OFF + NTO * KT * FRAG_SH) * 2;
constexpr int XW1_PITCH = 3 * NPG;                     // f32 per row
}

__device__ __forceinline__ unsigned short f2bf(float f) {
  unsigned u = __float_as_uint(f);
  u += 0x7FFFu + ((u >> 16) & 1u);        // round-to-nearest-even
  return (unsigned short)(u >> 16);
}
__device__ __forceinline__ float bf2f(unsigned short h) {
  return __uint_as_float(((unsigned)h) << 16);
}

__device__ __forceinline__ v8f wmma_bf16(v16bf a, v16bf b, v8f c) {
  return __builtin_amdgcn_wmma_f32_16x16x32_bf16(
      /*neg_a=*/false, a, /*neg_b=*/false, b,
      /*c_mod=*/(short)0, c, /*reuse_a=*/false, /*reuse_b=*/false);
}

// A-fragment (16x32 bf16) from a row-major [16][KP] bf16 LDS buffer.
// lane&15 = row M; lanes<16 hold K 0..15, lanes>=16 hold K 16..31.
__device__ __forceinline__ v16bf load_A(const unsigned short* buf, int kt, int lane) {
  const int row = lane & 15;
  const int kb  = kt * 32 + ((lane >> 4) << 4);
  const unsigned short* p = buf + row * KP + kb;
  v16bf a;
  ((v8bf*)&a)[0] = *(const v8bf*)(p);        // K kb..kb+7    (ds_load_b128)
  ((v8bf*)&a)[1] = *(const v8bf*)(p + 8);    // K kb+8..kb+15
  return a;
}

// B-fragment (32x16 bf16) pre-packed so each lane reads 32 contiguous bytes.
__device__ __forceinline__ v16bf load_B(const unsigned short* frag, int lane) {
  const unsigned short* p = frag + lane * 16;
  v16bf b;
  ((v8bf*)&b)[0] = *(const v8bf*)(p);        // global_load_b128
  ((v8bf*)&b)[1] = *(const v8bf*)(p + 8);
  return b;
}

// ---------------------------------------------------------------------------
// Weight packing: W[g*rowsPerGate+n][k] (f32) -> bf16 B-fragments.
// Fragment element: lane l, slot j -> K = (l>>4)*16 + j, N = l&15. Zero-padded.
// ---------------------------------------------------------------------------
__global__ void pack_weights(const float* __restrict__ W,
                             unsigned short* __restrict__ dst,
                             int gates, int ntiles, int rowsPerGate, int kdim) {
  const int idx = blockIdx.x * blockDim.x + threadIdx.x;
  const int total = gates * ntiles * KT * FRAG_SH;
  if (idx >= total) return;
  const int e  = idx & 511;
  const int f  = idx >> 9;
  const int kt = f % KT;
  const int nt = (f / KT) % ntiles;
  const int g  = f / (KT * ntiles);
  const int lane = e >> 4;
  const int j    = e & 15;
  const int kk = ((lane >> 4) << 4) + j;
  const int nn = lane & 15;
  const int n = nt * 16 + nn;
  const int k = kt * 32 + kk;
  float v = 0.f;
  if (n < rowsPerGate && k < kdim)
    v = W[((size_t)g * rowsPerGate + n) * kdim + k];
  dst[(size_t)f * FRAG_SH + e] = f2bf(v);
}

// ---------------------------------------------------------------------------
// One GRU layer step for this block's 16 batch rows; each wave takes every
// other n-tile. 6 register accumulators (r,z,n x input,hidden), fused
// elementwise update on D-layout, writes new h (bf16) to LDS double-buffer.
// ---------------------------------------------------------------------------
template <bool HAS_X>
__device__ __forceinline__ void gru_layer(
    int lane, int wave, int R, int col, int rb,
    const unsigned short* __restrict__ hp,   // prev h, LDS A-layout
    unsigned short* __restrict__ hn,         // new h, LDS A-layout
    const unsigned short* __restrict__ xs,   // layer input (LDS), HAS_X only
    const float* __restrict__ xw1g,          // precomputed xW1 (+bih1), !HAS_X
    const unsigned short* __restrict__ pWih, // packed Wih, HAS_X only
    const unsigned short* __restrict__ pWhh, // packed Whh
    const float* __restrict__ bih,           // HAS_X only
    const float* __restrict__ bhh,
    float* __restrict__ hd)                  // final-h global dest or nullptr
{
  for (int nt = wave; nt < NTG; nt += 2) {
    const int c = nt * 16 + col;
    v8f ax0 = {}, ax1 = {}, ax2 = {};
    v8f ah0 = {}, ah1 = {}, ah2 = {};
    if (!HAS_X) {
      const float* xp = xw1g + (size_t)(R + rb) * XW1_PITCH + c;
#pragma unroll
      for (int v = 0; v < 8; v++) {
        ax0[v] = xp[(size_t)v * XW1_PITCH];
        ax1[v] = xp[(size_t)v * XW1_PITCH + NPG];
        ax2[v] = xp[(size_t)v * XW1_PITCH + 2 * NPG];
      }
    }
    const unsigned short* wh = pWhh + (size_t)nt * KT * FRAG_SH;
    const unsigned short* wx = HAS_X ? (pWih + (size_t)nt * KT * FRAG_SH) : nullptr;
    __builtin_prefetch(wh + (size_t)KT * FRAG_SH, 0, 0);  // next tile's frags
    for (int kt = 0; kt < KT; kt++) {
      v16bf Ah = load_A(hp, kt, lane);
      ah0 = wmma_bf16(Ah, load_B(wh + (size_t)(0 * NTG * KT + kt) * FRAG_SH, lane), ah0);
      ah1 = wmma_bf16(Ah, load_B(wh + (size_t)(1 * NTG * KT + kt) * FRAG_SH, lane), ah1);
      ah2 = wmma_bf16(Ah, load_B(wh + (size_t)(2 * NTG * KT + kt) * FRAG_SH, lane), ah2);
      if (HAS_X) {
        v16bf Ax = load_A(xs, kt, lane);
        ax0 = wmma_bf16(Ax, load_B(wx + (size_t)(0 * NTG * KT + kt) * FRAG_SH, lane), ax0);
        ax1 = wmma_bf16(Ax, load_B(wx + (size_t)(1 * NTG * KT + kt) * FRAG_SH, lane), ax1);
        ax2 = wmma_bf16(Ax, load_B(wx + (size_t)(2 * NTG * KT + kt) * FRAG_SH, lane), ax2);
      }
    }
    if (c < H_) {
      float bxr = 0.f, bxz = 0.f, bxn = 0.f;
      if (HAS_X) { bxr = bih[c]; bxz = bih[H_ + c]; bxn = bih[2 * H_ + c]; }
      const float bhr = bhh[c], bhz = bhh[H_ + c], bhn = bhh[2 * H_ + c];
#pragma unroll
      for (int v = 0; v < 8; v++) {
        const int r = rb + v;
        const float rg = 1.f / (1.f + __expf(-((ax0[v] + bxr) + (ah0[v] + bhr))));
        const float zg = 1.f / (1.f + __expf(-((ax1[v] + bxz) + (ah1[v] + bhz))));
        const float ng = tanhf((ax2[v] + bxn) + rg * (ah2[v] + bhn));
        const float hprev = bf2f(hp[r * KP + c]);
        const float hv = (1.f - zg) * ng + zg * hprev;
        hn[r * KP + c] = f2bf(hv);
        if (hd) hd[(size_t)(R + r) * H_ + c] = hv;
      }
    }
  }
}

// ---------------------------------------------------------------------------
// Fused kernel: 128 blocks x 64 threads (2 waves); block owns 16 batch rows.
// ---------------------------------------------------------------------------
__global__ __launch_bounds__(64) void gru_fused(
    const float* __restrict__ enc,
    const float* __restrict__ h01, const float* __restrict__ h02,
    const float* __restrict__ h03,
    const float* __restrict__ gamma, const float* __restrict__ beta,
    const float* __restrict__ mean,  const float* __restrict__ var,
    const float* __restrict__ fc_in_b,
    const float* __restrict__ bhh1,
    const float* __restrict__ bih1,
    const float* __restrict__ bih2, const float* __restrict__ bhh2,
    const float* __restrict__ bih3, const float* __restrict__ bhh3,
    const float* __restrict__ fc_out_b,
    const unsigned short* __restrict__ pk,
    float* __restrict__ xw1,
    float* __restrict__ out)
{
  __shared__ __align__(16) unsigned short sbuf[6 * BUF_SH];  // 43008 B
  const int lane = threadIdx.x & 31;
  const int wave = threadIdx.x >> 5;
  const int R    = blockIdx.x * 16;
  const int col  = lane & 15;          // D-layout column within tile
  const int rb   = (lane >> 4) * 8;    // D-layout row base

  // Zero all h buffers (pads must stay zero forever).
  for (int i = threadIdx.x; i < 6 * BUF_SH; i += 64) sbuf[i] = 0;
  __syncthreads();

  // Stage h0 (bf16, A-layout) into buffers 0/2/4; stage bn(x) into buffer 1.
  for (int i = threadIdx.x; i < 16 * H_; i += 64) {
    const int r = i / H_, c = i % H_;
    sbuf[0 * BUF_SH + r * KP + c] = f2bf(h01[(size_t)(R + r) * H_ + c]);
    sbuf[2 * BUF_SH + r * KP + c] = f2bf(h02[(size_t)(R + r) * H_ + c]);
    sbuf[4 * BUF_SH + r * KP + c] = f2bf(h03[(size_t)(R + r) * H_ + c]);
  }
  for (int i = threadIdx.x; i < 16 * KP; i += 64) {
    const int r = i / KP, c = i % KP;
    float v = 0.f;
    if (c < I_) {
      const float x = enc[(size_t)(R + r) * I_ + c];
      v = (x - mean[c]) * rsqrtf(var[c] + 1e-5f) * gamma[c] + beta[c];
    }
    sbuf[1 * BUF_SH + r * KP + c] = f2bf(v);
  }
  __syncthreads();

  // emb = relu(bn(x) @ W_in^T + b)  -> buffer 3 (bf16)
  for (int nt = wave; nt < NTG; nt += 2) {
    v8f acc = {};
    for (int kt = 0; kt < KT; kt++) {
      v16bf a = load_A(sbuf + 1 * BUF_SH, kt, lane);
      acc = wmma_bf16(a, load_B(pk + WIN_OFF + (size_t)(nt * KT + kt) * FRAG_SH, lane), acc);
    }
    const int c = nt * 16 + col;
    if (c < H_) {
      const float bb = fc_in_b[c];
#pragma unroll
      for (int v = 0; v < 8; v++) {
        float e = acc[v] + bb;
        e = e > 0.f ? e : 0.f;
        sbuf[3 * BUF_SH + (rb + v) * KP + c] = f2bf(e);
      }
    }
  }
  __syncthreads();

  // xW1 = emb @ Wih1^T + bih1 (time-invariant) -> workspace f32 [row][3][208]
  for (int j = wave; j < 3 * NTG; j += 2) {
    const int g = j / NTG, nt = j % NTG;
    v8f acc = {};
    for (int kt = 0; kt < KT; kt++) {
      v16bf a = load_A(sbuf + 3 * BUF_SH, kt, lane);
      acc = wmma_bf16(a, load_B(pk + WIH1_OFF + (size_t)((g * NTG + nt) * KT + kt) * FRAG_SH, lane), acc);
    }
    const int c = nt * 16 + col;
    const float bb = (c < H_) ? bih1[g * H_ + c] : 0.f;
#pragma unroll
    for (int v = 0; v < 8; v++)
      xw1[(size_t)(R + rb + v) * XW1_PITCH + g * NPG + c] =
          (c < H_) ? (acc[v] + bb) : 0.f;
  }
  __syncthreads();

  const size_t OUT_LEN = (size_t)B_ * T_ * O_;
  float* hout1 = out + OUT_LEN;
  float* hout2 = hout1 + (size_t)B_ * H_;
  float* hout3 = hout2 + (size_t)B_ * H_;

  for (int t = 0; t < T_; t++) {
    const int pp = t & 1;  // ping-pong the h double-buffers
    unsigned short* h1p = sbuf + (0 + pp) * BUF_SH;
    unsigned short* h1n = sbuf + (1 - pp) * BUF_SH;
    unsigned short* h2p = sbuf + (2 + pp) * BUF_SH;
    unsigned short* h2n = sbuf + (3 - pp) * BUF_SH;
    unsigned short* h3p = sbuf + (4 + pp) * BUF_SH;
    unsigned short* h3n = sbuf + (5 - pp) * BUF_SH;
    const bool last = (t == T_ - 1);

    gru_layer<false>(lane, wave, R, col, rb, h1p, h1n, nullptr, xw1,
                     nullptr, pk + WHH1_OFF, nullptr, bhh1,
                     last ? hout1 : nullptr);
    __syncthreads();
    gru_layer<true>(lane, wave, R, col, rb, h2p, h2n, h1n, nullptr,
                    pk + WIH2_OFF, pk + WHH2_OFF, bih2, bhh2,
                    last ? hout2 : nullptr);
    __syncthreads();
    gru_layer<true>(lane, wave, R, col, rb, h3p, h3n, h2n, nullptr,
                    pk + WIH3_OFF, pk + WHH3_OFF, bih3, bhh3,
                    last ? hout3 : nullptr);
    __syncthreads();

    // out[:, t, :] = sigmoid(h3 @ W_out^T + b)   (relu(sigmoid) == sigmoid)
    for (int ot = wave; ot < NTO; ot += 2) {
      v8f acc = {};
      for (int kt = 0; kt < KT; kt++) {
        v16bf a = load_A(h3n, kt, lane);
        acc = wmma_bf16(a, load_B(pk + WOUT_OFF + (size_t)(ot * KT + kt) * FRAG_SH, lane), acc);
      }
      const int c = ot * 16 + col;
      if (c < O_) {
        const float bb = fc_out_b[c];
#pragma unroll
        for (int v = 0; v < 8; v++) {
          const float s = 1.f / (1.f + __expf(-(acc[v] + bb)));
          out[((size_t)(R + rb + v) * T_ + t) * O_ + c] = s;
        }
      }
    }
    __syncthreads();
  }
}

// ---------------------------------------------------------------------------
extern "C" void kernel_launch(void* const* d_in, const int* in_sizes, int n_in,
                              void* d_out, int out_size, void* d_ws, size_t ws_size,
                              hipStream_t stream) {
  (void)in_sizes; (void)n_in; (void)out_size; (void)ws_size;
  const float* enc      = (const float*)d_in[0];
  const float* h01      = (const float*)d_in[1];
  const float* h02      = (const float*)d_in[2];
  const float* h03      = (const float*)d_in[3];
  const float* bn_gamma = (const float*)d_in[4];
  const float* bn_beta  = (const float*)d_in[5];
  const float* bn_mean  = (const float*)d_in[6];
  const float* bn_var   = (const float*)d_in[7];
  const float* fc_in_W  = (const float*)d_in[8];
  const float* fc_in_b  = (const float*)d_in[9];
  const float* Wih1     = (const float*)d_in[10];
  const float* Whh1     = (const float*)d_in[11];
  const float* bih1     = (const float*)d_in[12];
  const float* bhh1     = (const float*)d_in[13];
  const float* Wih2     = (const float*)d_in[14];
  const float* Whh2     = (const float*)d_in[15];
  const float* bih2     = (const float*)d_in[16];
  const float* bhh2     = (const float*)d_in[17];
  const float* Wih3     = (const float*)d_in[18];
  const float* Whh3     = (const float*)d_in[19];
  const float* bih3     = (const float*)d_in[20];
  const float* bhh3     = (const float*)d_in[21];
  const float* fc_out_W = (const float*)d_in[22];
  const float* fc_out_b = (const float*)d_in[23];

  unsigned short* pk = (unsigned short*)d_ws;
  float* xw1 = (float*)((char*)d_ws + XW1_OFF_BYTES);
  float* out = (float*)d_out;

  auto packLaunch = [&](const float* W, int offSh, int gates, int ntiles, int rows) {
    const int total = gates * ntiles * KT * FRAG_SH;
    pack_weights<<<(total + 255) / 256, 256, 0, stream>>>(W, pk + offSh, gates,
                                                          ntiles, rows, 200);
  };
  packLaunch(fc_in_W,  WIN_OFF,  1, NTG, 200);
  packLaunch(Wih1,     WIH1_OFF, 3, NTG, 200);
  packLaunch(Whh1,     WHH1_OFF, 3, NTG, 200);
  packLaunch(Wih2,     WIH2_OFF, 3, NTG, 200);
  packLaunch(Whh2,     WHH2_OFF, 3, NTG, 200);
  packLaunch(Wih3,     WIH3_OFF, 3, NTG, 200);
  packLaunch(Whh3,     WHH3_OFF, 3, NTG, 200);
  packLaunch(fc_out_W, WOUT_OFF, 1, NTO, 53);

  gru_fused<<<B_ / 16, 64, 0, stream>>>(
      enc, h01, h02, h03, bn_gamma, bn_beta, bn_mean, bn_var, fc_in_b,
      bhh1, bih1, bih2, bhh2, bih3, bhh3, fc_out_b, pk, xw1, out);
}
